// TLModel_12232066859584
// MI455X (gfx1250) — compile-verified, matrix operand order checked by reading, not run on hardware
//
#include <hip/hip_runtime.h>
#include <hip/hip_bf16.h>

// ---- problem constants (from reference) ----
#define BSZ   4096          // batch
#define KDIM  22000         // C*T = 22*1000
#define KPAD  22016         // padded to multiple of 32 (WMMA bf16 K-step)
#define DDIM  1472          // backbone feature dim (92 * 16)
#define NHEAD 8
#define HDIM  128
#define ODIM  4

typedef __attribute__((ext_vector_type(16))) __bf16 bf16x16;
typedef __attribute__((ext_vector_type(8)))  __bf16 bf16x8;
typedef __attribute__((ext_vector_type(8)))  float  f32x8;

// ---------------------------------------------------------------------------
// 1) x [B][KDIM] f32  ->  Xb [B][KPAD] bf16 (zero-padded tail)
// grid = (KPAD/256, B), block = 256
// ---------------------------------------------------------------------------
__global__ __launch_bounds__(256) void cvt_x_bf16(const float* __restrict__ x,
                                                  __bf16* __restrict__ xb) {
    const int k   = blockIdx.x * 256 + threadIdx.x;
    const int row = blockIdx.y;
    float v = (k < KDIM) ? x[(size_t)row * KDIM + k] : 0.0f;
    xb[(size_t)row * KPAD + k] = (__bf16)v;
}

// ---------------------------------------------------------------------------
// 2) Wb [KDIM][DDIM] f32 -> WbT [DDIM][KPAD] bf16, LDS-tiled transpose.
// grid = (KPAD/32, DDIM/32), block = (32, 8)
// ---------------------------------------------------------------------------
__global__ __launch_bounds__(256) void transpose_w_bf16(const float* __restrict__ Wb,
                                                        __bf16* __restrict__ WbT) {
    __shared__ float tile[32][33];           // +1 to dodge bank conflicts
    const int k0 = blockIdx.x * 32;
    const int n0 = blockIdx.y * 32;
    const int tx = threadIdx.x;              // 0..31
    const int ty = threadIdx.y;              // 0..7
#pragma unroll
    for (int i = 0; i < 4; ++i) {
        int k = k0 + ty + 8 * i;
        tile[ty + 8 * i][tx] = (k < KDIM) ? Wb[(size_t)k * DDIM + (n0 + tx)] : 0.0f;
    }
    __syncthreads();
#pragma unroll
    for (int i = 0; i < 4; ++i) {
        int n = n0 + ty + 8 * i;
        WbT[(size_t)n * KPAD + (k0 + tx)] = (__bf16)tile[tx][ty + 8 * i];
    }
}

// ---------------------------------------------------------------------------
// 3) Backbone GEMM: feats[B][DDIM] = Xb @ WbT^T + bb, bf16 WMMA, f32 acc.
//    Wave tile = 64(M) x 32(N) = 4x2 grid of v_wmma_f32_16x16x32_bf16.
//    Block = 8 waves (4x2) -> 256(M) x 64(N).  grid = (B/256, DDIM/64).
// ---------------------------------------------------------------------------
__global__ __launch_bounds__(256) void gemm_bf16_wmma(const __bf16* __restrict__ Xb,
                                                      const __bf16* __restrict__ WbT,
                                                      const float*  __restrict__ bb,
                                                      float* __restrict__ feats) {
    const int lane  = threadIdx.x & 31;
    const int wave  = threadIdx.x >> 5;      // 0..7
    const int waveM = wave >> 1;             // 0..3
    const int waveN = wave & 1;              // 0..1
    const int m0 = blockIdx.x * 256 + waveM * 64;
    const int n0 = blockIdx.y * 64  + waveN * 32;

    const int lr   = lane & 15;              // row (A) / col (B) within 16-tile
    const int kSel = (lane >> 4) * 8;        // 0 for lanes 0-15, 8 for 16-31

    // per-lane K-major base pointers (ISA 7.12.2 16-bit A/B layouts)
    const __bf16* aBase[4];
#pragma unroll
    for (int i = 0; i < 4; ++i)
        aBase[i] = Xb + (size_t)(m0 + i * 16 + lr) * KPAD + kSel;
    const __bf16* bBase[2];
#pragma unroll
    for (int j = 0; j < 2; ++j)
        bBase[j] = WbT + (size_t)(n0 + j * 16 + lr) * KPAD + kSel;

    f32x8 acc[4][2] = {};

#pragma unroll 2
    for (int k = 0; k < KPAD; k += 32) {
        bf16x16 a[4];
#pragma unroll
        for (int i = 0; i < 4; ++i) {
            bf16x8 lo = *(const bf16x8*)(aBase[i] + k);        // K = kSel..kSel+7
            bf16x8 hi = *(const bf16x8*)(aBase[i] + k + 16);   // K = 16+kSel..23+kSel
            a[i] = __builtin_shufflevector(lo, hi, 0, 1, 2, 3, 4, 5, 6, 7,
                                                   8, 9, 10, 11, 12, 13, 14, 15);
        }
        bf16x16 b[2];
#pragma unroll
        for (int j = 0; j < 2; ++j) {
            bf16x8 lo = *(const bf16x8*)(bBase[j] + k);
            bf16x8 hi = *(const bf16x8*)(bBase[j] + k + 16);
            b[j] = __builtin_shufflevector(lo, hi, 0, 1, 2, 3, 4, 5, 6, 7,
                                                   8, 9, 10, 11, 12, 13, 14, 15);
        }
#pragma unroll
        for (int i = 0; i < 4; ++i)
#pragma unroll
            for (int j = 0; j < 2; ++j)
                acc[i][j] = __builtin_amdgcn_wmma_f32_16x16x32_bf16(
                    /*neg_a=*/false, a[i], /*neg_b=*/false, b[j],
                    /*c_mod=*/(short)0, acc[i][j],
                    /*reuse_a=*/false, /*reuse_b=*/false);
    }

    // epilogue: C/D layout — VGPR r: row = r + 8*(lane>=16), col = lane%16
    const int rowSel = (lane >> 4) * 8;
#pragma unroll
    for (int i = 0; i < 4; ++i) {
#pragma unroll
        for (int j = 0; j < 2; ++j) {
            const int col = n0 + j * 16 + lr;
            const float bias = bb[col];
#pragma unroll
            for (int r = 0; r < 8; ++r) {
                const int row = m0 + i * 16 + rowSel + r;
                feats[(size_t)row * DDIM + col] = acc[i][j][r] + bias;
            }
        }
    }
}

// ---------------------------------------------------------------------------
// 4) Routed heads: each block = one sample, only its own head is evaluated.
//    out[b] = relu(feats[b] @ W1[id] + b1[id]) @ W2[id] + b2[id]
//    grid = B, block = 128 (= HDIM)
// ---------------------------------------------------------------------------
__global__ __launch_bounds__(128) void heads_kernel(const float* __restrict__ feats,
                                                    const int*   __restrict__ ids,
                                                    const float* __restrict__ W1,
                                                    const float* __restrict__ b1,
                                                    const float* __restrict__ W2,
                                                    const float* __restrict__ b2,
                                                    float* __restrict__ out) {
    __shared__ float sf[DDIM];
    __shared__ float sh[HDIM];
    const int b  = blockIdx.x;
    const int t  = threadIdx.x;              // 0..127, one hidden unit each
    const int id = ids[b];

    for (int d = t; d < DDIM; d += 128)
        sf[d] = feats[(size_t)b * DDIM + d];
    __syncthreads();

    const float* w1 = W1 + (size_t)id * DDIM * HDIM;   // [D][H], col t coalesced
    float acc = b1[id * HDIM + t];
#pragma unroll 4
    for (int d = 0; d < DDIM; ++d)
        acc = fmaf(sf[d], w1[(size_t)d * HDIM + t], acc);
    sh[t] = fmaxf(acc, 0.0f);
    __syncthreads();

    if (t < ODIM) {
        const float* w2 = W2 + (size_t)id * HDIM * ODIM;
        float o = b2[id * ODIM + t];
#pragma unroll 8
        for (int h = 0; h < HDIM; ++h)
            o = fmaf(sh[h], w2[h * ODIM + t], o);
        out[(size_t)b * ODIM + t] = o;
    }
}

// ---------------------------------------------------------------------------
extern "C" void kernel_launch(void* const* d_in, const int* in_sizes, int n_in,
                              void* d_out, int out_size, void* d_ws, size_t ws_size,
                              hipStream_t stream) {
    const float* x   = (const float*)d_in[0];
    const int*   ids = (const int*)  d_in[1];
    const float* Wb  = (const float*)d_in[2];
    const float* bb  = (const float*)d_in[3];
    const float* W1  = (const float*)d_in[4];
    const float* b1  = (const float*)d_in[5];
    const float* W2  = (const float*)d_in[6];
    const float* b2  = (const float*)d_in[7];
    float* out = (float*)d_out;

    // workspace layout (all 16B aligned):
    //   Xb    : BSZ  * KPAD bf16  = 180,355,072 B
    //   WbT   : DDIM * KPAD bf16  =  64,815,104 B
    //   feats : BSZ  * DDIM f32   =  24,117,248 B
    char* ws = (char*)d_ws;
    __bf16* Xb    = (__bf16*)ws;
    __bf16* WbT   = (__bf16*)(ws + (size_t)BSZ * KPAD * sizeof(__bf16));
    float*  feats = (float*)(ws + (size_t)BSZ * KPAD * sizeof(__bf16)
                                + (size_t)DDIM * KPAD * sizeof(__bf16));

    cvt_x_bf16<<<dim3(KPAD / 256, BSZ), 256, 0, stream>>>(x, Xb);
    transpose_w_bf16<<<dim3(KPAD / 32, DDIM / 32), dim3(32, 8), 0, stream>>>(Wb, WbT);
    gemm_bf16_wmma<<<dim3(BSZ / 256, DDIM / 64), 256, 0, stream>>>(Xb, WbT, bb, feats);
    heads_kernel<<<BSZ, 128, 0, stream>>>(feats, ids, W1, b1, W2, b2, out);
}
